// MMDoubleStreamBlock_70669391889168
// MI455X (gfx1250) — matrix operand order repaired and checked by reading, not compiled, not run on hardware
//
#include <hip/hip_runtime.h>
#include <math.h>

// ---------------- Types for CDNA5 WMMA ----------------
typedef __bf16 bf16;
typedef __attribute__((ext_vector_type(16))) __bf16 v16bf;
typedef __attribute__((ext_vector_type(8)))  __bf16 bf16x8;
typedef __attribute__((ext_vector_type(2)))  __bf16 bf16x2;
typedef __attribute__((ext_vector_type(8)))  float  v8f;

#define WMMA_BF16(a, b, c) \
  __builtin_amdgcn_wmma_f32_16x16x32_bf16(false, (a), false, (b), (short)0, (c), false, false)

// ---------------- Model constants ----------------
#define S_IMG 2048
#define S_TXT 256
#define S_TOT 2304
#define HID   3072
#define NH    24
#define HD    128
#define MLPD  12288
#define QKVN  9216

// LDS row pads (elements). All keep rows bank-conflict-free, the bf16 ones also 16B-aligned:
//  72 bf16 = 144B (stride 36 dw), 136 bf16 = 272B (stride 68 dw),
//  40 bf16 =  80B (stride 20 dw), 33 f32 = 132B.
#define PAD_G  72
#define PAD_Q  136
#define PAD_V  40
#define PAD_S  33

// =====================================================================
// Fragment loaders (wave32, CDNA5 documented layouts, 16-bit operands)
// =====================================================================
// A matrix 16x32 (M x K), bf16, source row-major with leading dim ld.
__device__ inline v16bf load_frag_A(const bf16* a, int ld) {
  const int lane = threadIdx.x & 31;
  const int m    = lane & 15;
  const int half = lane >> 4;
  const bf16* row = a + m * ld;
  v16bf f;
#pragma unroll
  for (int i = 0; i < 8; ++i) {
    const int k0 = ((i >> 2) * 16) + half * 8 + (i & 3) * 2;
    bf16x2 p = *(const bf16x2*)(row + k0);
    f[2 * i]     = p[0];
    f[2 * i + 1] = p[1];
  }
  return f;
}

// B matrix 32x16 (K x N), bf16, source stored N-major (bt[n][k], leading dim ld)
__device__ inline v16bf load_frag_B(const bf16* bt, int ld) {
  const int lane = threadIdx.x & 31;
  const int n    = lane & 15;
  const int half = lane >> 4;
  const bf16* col = bt + n * ld;
  v16bf f;
#pragma unroll
  for (int i = 0; i < 8; ++i) {
    const int k0 = ((i >> 2) * 16) + half * 8 + (i & 3) * 2;
    bf16x2 p = *(const bf16x2*)(col + k0);
    f[2 * i]     = p[0];
    f[2 * i + 1] = p[1];
  }
  return f;
}

__device__ inline float gelu_tanh(float x) {
  return 0.5f * x * (1.0f + tanhf(0.7978845608028654f * (x + 0.044715f * x * x * x)));
}

// =====================================================================
// silu(vec)
// =====================================================================
__global__ void silu_kernel(const float* __restrict__ vec, float* __restrict__ sv, int n) {
  int i = blockIdx.x * blockDim.x + threadIdx.x;
  if (i < n) {
    float x = vec[i];
    sv[i] = x / (1.0f + __expf(-x));
  }
}

// =====================================================================
// modulation GEMV: out[j] = sum_i sv[i] * W[i*N + j] + b[j]
// =====================================================================
__global__ void mod_gemv_kernel(const float* __restrict__ sv,
                                const float* __restrict__ W,
                                const float* __restrict__ b,
                                float* __restrict__ out, int K, int N) {
  __shared__ float s[HID];
  for (int i = threadIdx.x; i < K; i += blockDim.x) s[i] = sv[i];
  __syncthreads();
  int j = blockIdx.x * blockDim.x + threadIdx.x;
  if (j >= N) return;
  float acc = b[j];
  for (int i = 0; i < K; ++i) acc += s[i] * W[(size_t)i * N + j];
  out[j] = acc;
}

// =====================================================================
// LayerNorm + modulate: out_bf16 = ln(x) * (1 + sc) + sh
// =====================================================================
__global__ void ln_mod_kernel(const float* __restrict__ X,
                              const float* __restrict__ sh,
                              const float* __restrict__ sc,
                              bf16* __restrict__ out, int N) {
  const int row = blockIdx.x;
  const float* x = X + (size_t)row * N;
  __shared__ float red[256];
  float s = 0.f, ss = 0.f;
  for (int i = threadIdx.x; i < N; i += 256) {
    float v = x[i];
    s += v;
    ss += v * v;
  }
  red[threadIdx.x] = s;
  __syncthreads();
  for (int st = 128; st > 0; st >>= 1) {
    if (threadIdx.x < st) red[threadIdx.x] += red[threadIdx.x + st];
    __syncthreads();
  }
  const float mean = red[0] / (float)N;
  __syncthreads();
  red[threadIdx.x] = ss;
  __syncthreads();
  for (int st = 128; st > 0; st >>= 1) {
    if (threadIdx.x < st) red[threadIdx.x] += red[threadIdx.x + st];
    __syncthreads();
  }
  const float var = red[0] / (float)N - mean * mean;
  const float r = rsqrtf(var + 1e-6f);
  for (int i = threadIdx.x; i < N; i += 256) {
    float v = (x[i] - mean) * r;
    out[(size_t)row * N + i] = (bf16)(v * (1.0f + sc[i]) + sh[i]);
  }
}

// =====================================================================
// Tiled bf16 WMMA GEMM: C[M,N] = A[M,K] @ W[K,N] (+bias, templated epilogue)
// Tile 64(M) x 256(N), K-step 64. Block 256 = 8 waves (4 m x 2 n),
// each wave -> 16x128 via 16 v_wmma_f32_16x16x32_bf16 per K step.
// EPI: 0 = f32 store (bias); 1 = gelu -> bf16; 2 = residual + gate -> f32.
// =====================================================================
template <int EPI>
__global__ void gemm_bf16_kernel(const bf16* __restrict__ A,
                                 const float* __restrict__ W,
                                 const float* __restrict__ bias,
                                 const float* __restrict__ residual,
                                 const float* __restrict__ gate,
                                 float* __restrict__ outF,
                                 bf16* __restrict__ outB,
                                 int M, int N, int Kdim) {
  __shared__ alignas(16) bf16 sA[64 * PAD_G];    // [m][k], padded rows
  __shared__ alignas(16) bf16 sB[256 * PAD_G];   // [n][k]  (N-major), padded rows

  const int tm0 = blockIdx.y * 64;
  const int tn0 = blockIdx.x * 256;
  const int t = threadIdx.x;
  const int w = t >> 5;
  const int wm = w & 3;   // 0..3 : 16-row band
  const int wn = w >> 2;  // 0..1 : 128-col band

  v8f acc[8] = {};

  for (int k0 = 0; k0 < Kdim; k0 += 64) {
    // stage A tile (64x64 bf16): two 16B loads per thread, reg-buffered
    {
      const int row = t >> 2;
      const int col = (t & 3) * 16;
      bf16x8 a0 = *(const bf16x8*)&A[(size_t)(tm0 + row) * Kdim + k0 + col];
      bf16x8 a1 = *(const bf16x8*)&A[(size_t)(tm0 + row) * Kdim + k0 + col + 8];
      *(bf16x8*)&sA[row * PAD_G + col]     = a0;
      *(bf16x8*)&sA[row * PAD_G + col + 8] = a1;
    }
    // stage W tile (64x256 f32 -> bf16 N-major): k-pairs packed to b32 stores
    {
#pragma unroll
      for (int jj = 0; jj < 8; ++jj) {
        const int g  = t + jj * 256;     // group id, 0..2047
        const int k  = (g >> 6) * 2;     // even k, 0..62
        const int n0 = (g & 63) * 4;     // 0..252
        float4 va = *(const float4*)&W[(size_t)(k0 + k) * N + tn0 + n0];
        float4 vb = *(const float4*)&W[(size_t)(k0 + k + 1) * N + tn0 + n0];
        bf16x2 p0; p0[0] = (bf16)va.x; p0[1] = (bf16)vb.x;
        bf16x2 p1; p1[0] = (bf16)va.y; p1[1] = (bf16)vb.y;
        bf16x2 p2; p2[0] = (bf16)va.z; p2[1] = (bf16)vb.z;
        bf16x2 p3; p3[0] = (bf16)va.w; p3[1] = (bf16)vb.w;
        *(bf16x2*)&sB[(n0 + 0) * PAD_G + k] = p0;
        *(bf16x2*)&sB[(n0 + 1) * PAD_G + k] = p1;
        *(bf16x2*)&sB[(n0 + 2) * PAD_G + k] = p2;
        *(bf16x2*)&sB[(n0 + 3) * PAD_G + k] = p3;
      }
    }
    // prefetch a slice of the next W tile while we compute on this one
    if (k0 + 64 < Kdim) {
      __builtin_prefetch(&W[(size_t)(k0 + 64 + (t >> 3)) * N + tn0 + (t & 7) * 32], 0, 0);
    }
    __syncthreads();

#pragma unroll
    for (int kc = 0; kc < 64; kc += 32) {
      v16bf af = load_frag_A(&sA[(wm * 16) * PAD_G + kc], PAD_G);
#pragma unroll
      for (int s = 0; s < 8; ++s) {
        v16bf bfr = load_frag_B(&sB[(wn * 128 + s * 16) * PAD_G + kc], PAD_G);
        acc[s] = WMMA_BF16(af, bfr, acc[s]);
      }
    }
    __syncthreads();
  }

  // epilogue (branch-free per template)
  const int lane = t & 31;
  const int nn = lane & 15;
  const int half = lane >> 4;
#pragma unroll
  for (int s = 0; s < 8; ++s) {
    const int gn = tn0 + wn * 128 + s * 16 + nn;
    const float bv = bias[gn];
    float gv = 0.f;
    if (EPI == 2) gv = gate[gn];
#pragma unroll
    for (int i = 0; i < 8; ++i) {
      const int gm = tm0 + wm * 16 + half * 8 + i;
      float v = acc[s][i] + bv;
      if (EPI == 0) {
        outF[(size_t)gm * N + gn] = v;
      } else if (EPI == 1) {
        outB[(size_t)gm * N + gn] = (bf16)gelu_tanh(v);
      } else {
        outF[(size_t)gm * N + gn] = residual[(size_t)gm * N + gn] + gv * v;
      }
    }
  }
}

// =====================================================================
// QKV post-processing: per (row, head): RMS-norm q/k, optional RoPE,
// write Q/K/V bf16 head-major: [NH][S_TOT][HD]. Block = 64 threads.
// =====================================================================
__global__ void qkv_post_kernel(const float* __restrict__ qkv,
                                const float* __restrict__ qw,
                                const float* __restrict__ kw,
                                const float* __restrict__ cosT,
                                const float* __restrict__ sinT,
                                bf16* __restrict__ Qb,
                                bf16* __restrict__ Kb,
                                bf16* __restrict__ Vb,
                                int row_off, int use_rope) {
  const int s = blockIdx.x;
  const int h = blockIdx.y;
  const int t = threadIdx.x;  // 0..63
  const float* base = qkv + (size_t)s * QKVN;
  const float* q = base + h * HD;
  const float* k = base + HID + h * HD;
  const float* v = base + 2 * HID + h * HD;

  float q0 = q[2 * t], q1 = q[2 * t + 1];
  float k0 = k[2 * t], k1 = k[2 * t + 1];

  __shared__ float redq[64], redk[64];
  redq[t] = q0 * q0 + q1 * q1;
  redk[t] = k0 * k0 + k1 * k1;
  __syncthreads();
  for (int st = 32; st > 0; st >>= 1) {
    if (t < st) {
      redq[t] += redq[t + st];
      redk[t] += redk[t + st];
    }
    __syncthreads();
  }
  const float rq = rsqrtf(redq[0] / (float)HD + 1e-6f);
  const float rk = rsqrtf(redk[0] / (float)HD + 1e-6f);
  q0 = q0 * rq * qw[2 * t]; q1 = q1 * rq * qw[2 * t + 1];
  k0 = k0 * rk * kw[2 * t]; k1 = k1 * rk * kw[2 * t + 1];

  if (use_rope) {
    const float c = cosT[(size_t)s * (HD / 2) + t];
    const float sn = sinT[(size_t)s * (HD / 2) + t];
    float a = q0, b = q1;
    q0 = a * c - b * sn; q1 = b * c + a * sn;
    a = k0; b = k1;
    k0 = a * c - b * sn; k1 = b * c + a * sn;
  }
  const size_t o = ((size_t)h * S_TOT + row_off + s) * HD + 2 * t;
  Qb[o] = (bf16)q0; Qb[o + 1] = (bf16)q1;
  Kb[o] = (bf16)k0; Kb[o + 1] = (bf16)k1;
  Vb[o] = (bf16)v[2 * t]; Vb[o + 1] = (bf16)v[2 * t + 1];
}

// =====================================================================
// Flash attention. Grid: (S_TOT/64 q-tiles, NH heads). Block 256 = 8 waves.
// =====================================================================
__global__ void flash_attn_kernel(const bf16* __restrict__ Qb,
                                  const bf16* __restrict__ Kb,
                                  const bf16* __restrict__ Vb,
                                  bf16* __restrict__ attn_img,
                                  bf16* __restrict__ attn_txt,
                                  float scale) {
  __shared__ alignas(16) bf16 sQ[64 * PAD_Q];   // [q][d]
  __shared__ alignas(16) bf16 sK[32 * PAD_Q];   // [key][d]  (== B^T for QK^T)
  __shared__ alignas(16) bf16 sVt[128 * PAD_V]; // [d][key]  (== B^T for PV)
  __shared__ alignas(16) float sS[64 * PAD_S];
  __shared__ alignas(16) bf16 sP[64 * PAD_V];
  __shared__ float sM[64], sL[64], sAl[64];

  const int h = blockIdx.y;
  const int q0 = blockIdx.x * 64;
  const int t = threadIdx.x;
  const int w = t >> 5;
  const int lane = t & 31;
  const int wm = w & 3;   // row band (16 rows)
  const int wn = w >> 2;  // 0..1
  const int nn = lane & 15;
  const int half = lane >> 4;

  // load Q tile (64x128 bf16)
  const bf16* Qg = Qb + ((size_t)h * S_TOT + q0) * HD;
#pragma unroll
  for (int j = 0; j < 4; ++j) {
    int lin = t * 32 + j * 8;
    int r = lin >> 7, d = lin & 127;
    *(bf16x8*)&sQ[r * PAD_Q + d] = *(const bf16x8*)&Qg[(size_t)r * 128 + d];
  }
  if (t < 64) {
    sM[t] = -3.0e38f;
    sL[t] = 0.f;
  }
  v8f acc[4] = {};
  __syncthreads();

  for (int kt = 0; kt < S_TOT; kt += 32) {
    // stage K (32x128) and V^T (128x32) — register-buffered for MLP overlap
    {
      bf16x8 kreg[2], vreg[2];
#pragma unroll
      for (int j = 0; j < 2; ++j) {
        int lin = t * 16 + j * 8;
        int r = lin >> 7, d = lin & 127;
        const size_t gk = ((size_t)h * S_TOT + kt + r) * HD + d;
        kreg[j] = *(const bf16x8*)&Kb[gk];
        vreg[j] = *(const bf16x8*)&Vb[gk];
      }
#pragma unroll
      for (int j = 0; j < 2; ++j) {
        int lin = t * 16 + j * 8;
        int r = lin >> 7, d = lin & 127;
        *(bf16x8*)&sK[r * PAD_Q + d] = kreg[j];
#pragma unroll
        for (int e = 0; e < 8; ++e) sVt[(d + e) * PAD_V + r] = vreg[j][e];
      }
    }
    // prefetch next K/V tiles
    if (kt + 32 < S_TOT) {
      const size_t gn = ((size_t)h * S_TOT + kt + 32 + (t >> 3)) * HD + (t & 7) * 16;
      __builtin_prefetch(&Kb[gn], 0, 0);
      __builtin_prefetch(&Vb[gn], 0, 0);
    }
    __syncthreads();

    // S(64x32) = Q @ K^T * scale  — each wave a 16x16 subtile, K-dim = HD
    {
      v8f sacc = {};
#pragma unroll
      for (int kk = 0; kk < HD; kk += 32) {
        v16bf a = load_frag_A(&sQ[(wm * 16) * PAD_Q + kk], PAD_Q);
        v16bf b = load_frag_B(&sK[(wn * 16) * PAD_Q + kk], PAD_Q);
        sacc = WMMA_BF16(a, b, sacc);
      }
#pragma unroll
      for (int i = 0; i < 8; ++i) {
        const int m = wm * 16 + half * 8 + i;
        const int n = wn * 16 + nn;
        sS[m * PAD_S + n] = sacc[i] * scale;
      }
    }
    __syncthreads();

    // online softmax (one thread per row)
    if (t < 64) {
      float mx = sS[t * PAD_S];
#pragma unroll
      for (int j = 1; j < 32; ++j) mx = fmaxf(mx, sS[t * PAD_S + j]);
      const float mNew = fmaxf(sM[t], mx);
      const float alpha = __expf(sM[t] - mNew);
      float sum = 0.f;
#pragma unroll
      for (int j = 0; j < 32; ++j) {
        float p = __expf(sS[t * PAD_S + j] - mNew);
        sum += p;
        sP[t * PAD_V + j] = (bf16)p;
      }
      sL[t] = sL[t] * alpha + sum;
      sM[t] = mNew;
      sAl[t] = alpha;
    }
    __syncthreads();

    // O(64x128): wave owns rows wm*16..+16, cols wn*64..+64
    {
      v16bf a = load_frag_A(&sP[(wm * 16) * PAD_V], PAD_V);
#pragma unroll
      for (int s2 = 0; s2 < 4; ++s2) {
#pragma unroll
        for (int i = 0; i < 8; ++i)
          acc[s2][i] *= sAl[wm * 16 + half * 8 + i];
        v16bf b = load_frag_B(&sVt[(wn * 64 + s2 * 16) * PAD_V], PAD_V);
        acc[s2] = WMMA_BF16(a, b, acc[s2]);
      }
    }
    __syncthreads();
  }

  // finalize + scatter into [S, HID] bf16 (head-major within row)
#pragma unroll
  for (int s2 = 0; s2 < 4; ++s2) {
#pragma unroll
    for (int i = 0; i < 8; ++i) {
      const int m = wm * 16 + half * 8 + i;
      const int d = wn * 64 + s2 * 16 + nn;
      const int g = q0 + m;
      const float v = acc[s2][i] / sL[m];
      if (g < S_IMG)
        attn_img[(size_t)g * HID + h * HD + d] = (bf16)v;
      else
        attn_txt[(size_t)(g - S_IMG) * HID + h * HD + d] = (bf16)v;
    }
  }
}

// =====================================================================
// Host-side orchestration
// =====================================================================
extern "C" void kernel_launch(void* const* d_in, const int* in_sizes, int n_in,
                              void* d_out, int out_size, void* d_ws, size_t ws_size,
                              hipStream_t stream) {
  (void)in_sizes; (void)n_in; (void)out_size; (void)ws_size;

  const float* img        = (const float*)d_in[0];
  const float* txt        = (const float*)d_in[1];
  const float* vec        = (const float*)d_in[2];
  const float* cosT       = (const float*)d_in[3];
  const float* sinT       = (const float*)d_in[4];
  const float* img_mod_w  = (const float*)d_in[5];
  const float* img_mod_b  = (const float*)d_in[6];
  const float* img_qkv_w  = (const float*)d_in[7];
  const float* img_qkv_b  = (const float*)d_in[8];
  const float* img_qn_w   = (const float*)d_in[9];
  const float* img_kn_w   = (const float*)d_in[10];
  const float* img_proj_w = (const float*)d_in[11];
  const float* img_proj_b = (const float*)d_in[12];
  const float* img_mlp_w1 = (const float*)d_in[13];
  const float* img_mlp_b1 = (const float*)d_in[14];
  const float* img_mlp_w2 = (const float*)d_in[15];
  const float* img_mlp_b2 = (const float*)d_in[16];
  const float* txt_mod_w  = (const float*)d_in[17];
  const float* txt_mod_b  = (const float*)d_in[18];
  const float* txt_qkv_w  = (const float*)d_in[19];
  const float* txt_qkv_b  = (const float*)d_in[20];
  const float* txt_qn_w   = (const float*)d_in[21];
  const float* txt_kn_w   = (const float*)d_in[22];
  const float* txt_proj_w = (const float*)d_in[23];
  const float* txt_proj_b = (const float*)d_in[24];
  const float* txt_mlp_w1 = (const float*)d_in[25];
  const float* txt_mlp_b1 = (const float*)d_in[26];
  const float* txt_mlp_w2 = (const float*)d_in[27];
  const float* txt_mlp_b2 = (const float*)d_in[28];

  float* out_img = (float*)d_out;
  float* out_txt = (float*)d_out + (size_t)S_IMG * HID;

  // workspace carve-out
  char* ws = (char*)d_ws;
  size_t off = 0;
  auto carve = [&](size_t bytes) -> void* {
    void* p = ws + off;
    off += (bytes + 255) & ~(size_t)255;
    return p;
  };
  float* sv     = (float*)carve(HID * 4);
  float* modI   = (float*)carve(6 * HID * 4);
  float* modT   = (float*)carve(6 * HID * 4);
  bf16* xmodI   = (bf16*)carve((size_t)S_IMG * HID * 2);
  bf16* xmodT   = (bf16*)carve((size_t)S_TXT * HID * 2);
  float* qkvI   = (float*)carve((size_t)S_IMG * QKVN * 4);
  float* qkvT   = (float*)carve((size_t)S_TXT * QKVN * 4);
  bf16* Qb      = (bf16*)carve((size_t)NH * S_TOT * HD * 2);
  bf16* Kb      = (bf16*)carve((size_t)NH * S_TOT * HD * 2);
  bf16* Vb      = (bf16*)carve((size_t)NH * S_TOT * HD * 2);
  bf16* attnI   = (bf16*)carve((size_t)S_IMG * HID * 2);
  bf16* attnT   = (bf16*)carve((size_t)S_TXT * HID * 2);
  float* resI   = (float*)carve((size_t)S_IMG * HID * 4);
  float* resT   = (float*)carve((size_t)S_TXT * HID * 4);
  bf16* hI      = (bf16*)carve((size_t)S_IMG * MLPD * 2);
  bf16* hT      = (bf16*)carve((size_t)S_TXT * MLPD * 2);
  bf16* ln2I    = (bf16*)carve((size_t)S_IMG * HID * 2);
  bf16* ln2T    = (bf16*)carve((size_t)S_TXT * HID * 2);

  // 1) silu(vec)
  silu_kernel<<<HID / 256, 256, 0, stream>>>(vec, sv, HID);
  // 2) modulation vectors (6*HID each)
  mod_gemv_kernel<<<(6 * HID) / 256, 256, 0, stream>>>(sv, img_mod_w, img_mod_b, modI, HID, 6 * HID);
  mod_gemv_kernel<<<(6 * HID) / 256, 256, 0, stream>>>(sv, txt_mod_w, txt_mod_b, modT, HID, 6 * HID);
  // 3) ln + modulate (attn branch): sh = mod[0], sc = mod[1]
  ln_mod_kernel<<<S_IMG, 256, 0, stream>>>(img, modI, modI + HID, xmodI, HID);
  ln_mod_kernel<<<S_TXT, 256, 0, stream>>>(txt, modT, modT + HID, xmodT, HID);
  // 4) QKV GEMMs (EPI 0: bias -> f32)
  gemm_bf16_kernel<0><<<dim3(QKVN / 256, S_IMG / 64), 256, 0, stream>>>(
      xmodI, img_qkv_w, img_qkv_b, nullptr, nullptr, qkvI, nullptr, S_IMG, QKVN, HID);
  gemm_bf16_kernel<0><<<dim3(QKVN / 256, S_TXT / 64), 256, 0, stream>>>(
      xmodT, txt_qkv_w, txt_qkv_b, nullptr, nullptr, qkvT, nullptr, S_TXT, QKVN, HID);
  // 5) RMS norm + RoPE -> head-major bf16 Q/K/V
  qkv_post_kernel<<<dim3(S_IMG, NH), 64, 0, stream>>>(
      qkvI, img_qn_w, img_kn_w, cosT, sinT, Qb, Kb, Vb, 0, 1);
  qkv_post_kernel<<<dim3(S_TXT, NH), 64, 0, stream>>>(
      qkvT, txt_qn_w, txt_kn_w, cosT, sinT, Qb, Kb, Vb, S_IMG, 0);
  // 6) flash attention
  flash_attn_kernel<<<dim3(S_TOT / 64, NH), 256, 0, stream>>>(
      Qb, Kb, Vb, attnI, attnT, 0.08838834764831845f /* HD^-0.5 */);
  // 7) attention projection + gated residual (EPI 2)
  gemm_bf16_kernel<2><<<dim3(HID / 256, S_IMG / 64), 256, 0, stream>>>(
      attnI, img_proj_w, img_proj_b, img, modI + 2 * HID, resI, nullptr, S_IMG, HID, HID);
  gemm_bf16_kernel<2><<<dim3(HID / 256, S_TXT / 64), 256, 0, stream>>>(
      attnT, txt_proj_w, txt_proj_b, txt, modT + 2 * HID, resT, nullptr, S_TXT, HID, HID);
  // 8) ln + modulate (mlp branch): sh = mod[3], sc = mod[4]
  ln_mod_kernel<<<S_IMG, 256, 0, stream>>>(resI, modI + 3 * HID, modI + 4 * HID, ln2I, HID);
  ln_mod_kernel<<<S_TXT, 256, 0, stream>>>(resT, modT + 3 * HID, modT + 4 * HID, ln2T, HID);
  // 9) MLP up + GELU (EPI 1: bf16 out)
  gemm_bf16_kernel<1><<<dim3(MLPD / 256, S_IMG / 64), 256, 0, stream>>>(
      ln2I, img_mlp_w1, img_mlp_b1, nullptr, nullptr, nullptr, hI, S_IMG, MLPD, HID);
  gemm_bf16_kernel<1><<<dim3(MLPD / 256, S_TXT / 64), 256, 0, stream>>>(
      ln2T, txt_mlp_w1, txt_mlp_b1, nullptr, nullptr, nullptr, hT, S_TXT, MLPD, HID);
  // 10) MLP down + gated residual -> final outputs (EPI 2)
  gemm_bf16_kernel<2><<<dim3(HID / 256, S_IMG / 64), 256, 0, stream>>>(
      hI, img_mlp_w2, img_mlp_b2, resI, modI + 5 * HID, out_img, nullptr, S_IMG, HID, MLPD);
  gemm_bf16_kernel<2><<<dim3(HID / 256, S_TXT / 64), 256, 0, stream>>>(
      hT, txt_mlp_w2, txt_mlp_b2, resT, modT + 5 * HID, out_txt, nullptr, S_TXT, HID, MLPD);
}
